// QConvIntNet_20959440404723
// MI455X (gfx1250) — matrix operand order repaired and checked by reading, not compiled, not run on hardware
//
#include <hip/hip_runtime.h>
#include <math.h>

typedef __attribute__((ext_vector_type(16))) _Float16 v16h;
typedef __attribute__((ext_vector_type(8)))  float    v8f;
typedef __attribute__((ext_vector_type(4)))  int      v4i;

#define NCONST 64
#define NEDGES (NCONST * (NCONST - 1))
#define NFEATS 16
#define NTILES (NEDGES / 16)   // 252

// Guarded gfx1250 async-load-to-LDS path (falls back to plain copies).
#if defined(__HIP_DEVICE_COMPILE__) && \
    __has_builtin(__builtin_amdgcn_global_load_async_to_lds_b128) && \
    __has_builtin(__builtin_amdgcn_s_wait_asynccnt)
#define HAVE_ASYNC 1
typedef __attribute__((address_space(1))) v4i gv4i;   // global int4
typedef __attribute__((address_space(3))) v4i lv4i;   // LDS int4
#else
#define HAVE_ASYNC 0
#endif

// qkeras quantized_bits(8,0): clip(round(w*128)/128, -1, 127/128)
__device__ __forceinline__ float qb(float w) {
    float q = rintf(w * 128.f) * (1.f / 128.f);
    return fminf(fmaxf(q, -1.f), 127.f / 128.f);
}
// qkeras quantized_relu(8,0): clip(round(relu(x)*256)/256, 0, 255/256)
__device__ __forceinline__ float qrelu8(float x) {
    float r = fmaxf(x, 0.f);
    float q = rintf(r * 256.f) * (1.f / 256.f);
    return fminf(q, 255.f / 256.f);
}

union ABReg { v16h v; unsigned u[8]; _Float16 h[16]; };

__device__ __forceinline__ unsigned packh2(float a, float b) {
    union { _Float16 h[2]; unsigned u; } p;
    p.h[0] = (_Float16)a; p.h[1] = (_Float16)b;
    return p.u;
}

__device__ __forceinline__ v8f wmma16(const ABReg& a, const ABReg& b, v8f c) {
    return __builtin_amdgcn_wmma_f32_16x16x32_f16(
        /*neg_a=*/false, a.v, /*neg_b=*/false, b.v,
        /*c_mod=*/(short)0, c, /*reuse_a=*/false, /*reuse_b=*/false);
}

// ---------------------------------------------------------------------------
// Weight prep: quantize + pack into WMMA B-matrix "register images".
// B layout (16-bit, 32x16): VGPR j, lane l -> N = l%16,
//   K = 2j + halfIdx + (j>=4 ? 8 : 0) + (l>=16 ? 8 : 0); low half = even K.
// ws layout (u32 units): images 0..11 at [img*256 .. +255]; biases (f32,
// quantized, zero-padded) at float index 3072..3231.
// ---------------------------------------------------------------------------
__global__ void qin_prep(const float* fr_w1, const float* fr_w2, const float* fr_w3,
                         const float* fo_w1, const float* fo_w2, const float* fo_w3,
                         const float* fr_b1, const float* fr_b2, const float* fr_b3,
                         const float* fo_b1, const float* fo_b2, const float* fo_b3,
                         unsigned* wsu, float* wsf) {
    const int img = blockIdx.x;
    const int t = threadIdx.x;
    if (img < 12) {
        const float* W; int K, N, kb, nb;
        switch (img) {
            case 0:  W = fr_w1; K = 32; N = 30; kb = 0;  nb = 0;  break; // f_r L1, N 0-15
            case 1:  W = fr_w1; K = 32; N = 30; kb = 0;  nb = 16; break; // f_r L1, N 16-29
            case 2:  W = fr_w2; K = 30; N = 15; kb = 0;  nb = 0;  break; // f_r L2
            case 3:  W = fr_w3; K = 15; N = 6;  kb = 0;  nb = 0;  break; // f_r L3
            case 4:  W = fo_w1; K = 22; N = 45; kb = 0;  nb = 0;  break; // f_o L1, N 0-15
            case 5:  W = fo_w1; K = 22; N = 45; kb = 0;  nb = 16; break; // f_o L1, N 16-31
            case 6:  W = fo_w1; K = 22; N = 45; kb = 0;  nb = 32; break; // f_o L1, N 32-44
            case 7:  W = fo_w2; K = 45; N = 22; kb = 0;  nb = 0;  break; // f_o L2, K0, N0
            case 8:  W = fo_w2; K = 45; N = 22; kb = 0;  nb = 16; break; // f_o L2, K0, N1
            case 9:  W = fo_w2; K = 45; N = 22; kb = 32; nb = 0;  break; // f_o L2, K1, N0
            case 10: W = fo_w2; K = 45; N = 22; kb = 32; nb = 16; break; // f_o L2, K1, N1
            default: W = fo_w3; K = 22; N = 6;  kb = 0;  nb = 0;  break; // f_o L3
        }
        const int v = t >> 5, lane = t & 31;
        const int n = nb + (lane & 15);
        const int k0 = kb + 2 * v + ((v >= 4) ? 8 : 0) + ((lane >= 16) ? 8 : 0);
        float w0 = (k0     < K && n < N) ? qb(W[k0 * N + n])       : 0.f;
        float w1 = (k0 + 1 < K && n < N) ? qb(W[(k0 + 1) * N + n]) : 0.f;
        wsu[img * 256 + t] = packh2(w0, w1);
    } else if (t < 160) {
        // padded quantized biases: frb1@0(32) frb2@32(16) frb3@48(16)
        //                          fob1@64(48) fob2@112(32) fob3@144(16)
        float val = 0.f;
        if (t < 32)       { if (t < 30) val = qb(fr_b1[t]); }
        else if (t < 48)  { int i = t - 32;  if (i < 15) val = qb(fr_b2[i]); }
        else if (t < 64)  { int i = t - 48;  if (i < 6)  val = qb(fr_b3[i]); }
        else if (t < 112) { int i = t - 64;  if (i < 45) val = qb(fo_b1[i]); }
        else if (t < 144) { int i = t - 112; if (i < 22) val = qb(fo_b2[i]); }
        else              { int i = t - 144; if (i < 6)  val = qb(fo_b3[i]); }
        wsf[3072 + t] = val;
    }
}

// ---------------------------------------------------------------------------
// Main fused kernel: one block (8 wave32) per jet.
// ---------------------------------------------------------------------------
__global__ __launch_bounds__(256) void qin_main(
    const float* __restrict__ x,
    const float* __restrict__ bn_gamma, const float* __restrict__ bn_beta,
    const float* __restrict__ bn_mean,  const float* __restrict__ bn_var,
    const float* __restrict__ fc_w1, const float* __restrict__ fc_b1,
    const float* __restrict__ fc_w2, const float* __restrict__ fc_b2,
    const unsigned* __restrict__ wsu, const float* __restrict__ wsf,
    float* __restrict__ out)
{
    __shared__ __align__(16) unsigned Bimg[12 * 256];    // weight reg images
    __shared__ __align__(16) float    xstage[NCONST * NFEATS]; // raw x tile
    __shared__ unsigned normH[NCONST * 8];    // 64 x 16 f16 BN output
    __shared__ unsigned ceH[NCONST * 16];     // 64 x 32 f16 concat(norm, ebar, 0)
    __shared__ float    ebar[NCONST * 6];     // aggregated effects
    __shared__ unsigned scr[8][512];          // per-wave 16x64 f16 relayout scratch
    __shared__ float    biasL[160];
    __shared__ float    dynL[8];
    __shared__ float    afc[48];
    __shared__ float    logitsL[5];

    const int tid  = threadIdx.x;
    const int lane = tid & 31;
    const int w    = tid >> 5;                 // wave id 0..7
    const int b    = blockIdx.x;
    const int m16  = lane & 15;
    const int hiM  = (lane >= 16) ? 8 : 0;     // C/D: upper lanes hold M+8
    const int hi4  = (lane >= 16) ? 4 : 0;     // A: upper lanes K+8 -> +4 dwords

    const float* gx = x + b * NCONST * NFEATS;

    // ---- stage weights + x tile into LDS (async path if available) ----
#if HAVE_ASYNC
    {
        #pragma unroll
        for (int k = 0; k < 3; ++k) {
            int i = (tid + k * 256) * 4;       // dword index, 16B granules
            __builtin_amdgcn_global_load_async_to_lds_b128(
                (gv4i*)(wsu + i), (lv4i*)(&Bimg[i]), 0, 0);
        }
        __builtin_amdgcn_global_load_async_to_lds_b128(
            (gv4i*)(gx + tid * 4), (lv4i*)(&xstage[tid * 4]), 0, 0);
        __builtin_amdgcn_s_wait_asynccnt(0);
    }
#else
    for (int i = tid; i < 12 * 256; i += 256) Bimg[i] = wsu[i];
    for (int i = tid; i < NCONST * NFEATS; i += 256) xstage[i] = gx[i];
#endif
    for (int i = tid; i < 160; i += 256)        biasL[i] = wsf[3072 + i];
    for (int i = tid; i < NCONST * 6; i += 256) ebar[i] = 0.f;
    if (tid < 8) dynL[tid] = 0.f;
    __syncthreads();

    // ---- BatchNorm -> f16 norm in LDS ----
    _Float16* nh = (_Float16*)normH;
    for (int i = tid; i < NCONST * NFEATS; i += 256) {
        int f = i & 15;
        float v = (xstage[i] - bn_mean[f]) *
                  rsqrtf(bn_var[f] + 1e-3f) * bn_gamma[f] + bn_beta[f];
        nh[i] = (_Float16)v;
    }
    __syncthreads();

    // ======================= edge stage: f_r over 4032 edges ================
    ABReg b10, b11, b2r, b3r;
    #pragma unroll
    for (int v = 0; v < 8; ++v) {
        b10.u[v] = Bimg[0 * 256 + v * 32 + lane];
        b11.u[v] = Bimg[1 * 256 + v * 32 + lane];
        b2r.u[v] = Bimg[2 * 256 + v * 32 + lane];
        b3r.u[v] = Bimg[3 * 256 + v * 32 + lane];
    }
    _Float16* sh  = (_Float16*)&scr[w][0];
    unsigned* shu = &scr[w][0];
    const float bb10 = biasL[m16];        // fr_b1[n], fr_b1[16+n]
    const float bb11 = biasL[16 + m16];
    const float bb2  = biasL[32 + m16];   // fr_b2
    const float bb3  = biasL[48 + m16];   // fr_b3

    // Wave-uniform receiver bookkeeping: edge ebase = tile*16, receiver r0 =
    // ebase/63, offset j0 = ebase%63 kept in SGPRs; a 16-edge tile spans at
    // most two receivers, and tile advances by 128 edges = 2*63 + 2 per
    // iteration. This removes all integer division from the hot loop.
    int r0 = __builtin_amdgcn_readfirstlane((w * 16) / 63);
    int j0 = __builtin_amdgcn_readfirstlane((w * 16) % 63);

    for (int tile = w; tile < NTILES; tile += 8) {
        // ---- A-row edge decode (division-free) ----
        int jm   = j0 + m16;
        int wrap = (jm >= 63) ? 1 : 0;
        int r    = r0 + wrap;
        int jj   = jm - (wrap ? 63 : 0);
        int s    = jj + ((jj >= r) ? 1 : 0);
        ABReg a;
        #pragma unroll
        for (int v = 0; v < 4; ++v) {
            a.u[v]     = normH[r * 8 + v + hi4];   // K 0..15  -> receiver feats
            a.u[v + 4] = normH[s * 8 + v + hi4];   // K 16..31 -> sender feats
        }
        // ---- layer 1: 32 -> 30 (two N-tiles) ----
        v8f c0 = {}, c1 = {};
        c0 = wmma16(a, b10, c0);
        c1 = wmma16(a, b11, c1);
        #pragma unroll
        for (int v = 0; v < 8; ++v) {
            int M = v + hiM;
            sh[M * 32 + m16]      = (_Float16)qrelu8(c0[v] + bb10);
            sh[M * 32 + 16 + m16] = (_Float16)qrelu8(c1[v] + bb11);
        }
        // ---- layer 2: 30 -> 15 ----
        ABReg a2;
        #pragma unroll
        for (int v = 0; v < 4; ++v) {
            a2.u[v]     = shu[m16 * 16 + v + hi4];
            a2.u[v + 4] = shu[m16 * 16 + 8 + v + hi4];
        }
        v8f c2 = {};
        c2 = wmma16(a2, b2r, c2);
        #pragma unroll
        for (int v = 0; v < 8; ++v)
            sh[(v + hiM) * 32 + m16] = (_Float16)qrelu8(c2[v] + bb2);
        // ---- layer 3: 15 -> 6, qrelu, scatter-add into ebar ----
        ABReg a3;
        #pragma unroll
        for (int v = 0; v < 4; ++v) {
            a3.u[v]     = shu[m16 * 16 + v + hi4];
            a3.u[v + 4] = shu[m16 * 16 + 8 + v + hi4];
        }
        v8f c3 = {};
        c3 = wmma16(a3, b3r, c3);
        if (m16 < 6) {
            int base6 = r0 * 6 + m16;
            #pragma unroll
            for (int v = 0; v < 8; ++v) {
                int idx = base6 + (((j0 + v + hiM) >= 63) ? 6 : 0);
                atomicAdd(&ebar[idx], qrelu8(c3[v] + bb3));
            }
        }
        // advance 128 edges: r0 += 2, j0 += 2 (with single wrap)
        j0 += 2; r0 += 2;
        if (j0 >= 63) { j0 -= 63; r0 += 1; }
    }
    __syncthreads();

    // ---- build ce = concat(norm, ebar) as f16 (64 x 32, zero-padded) ----
    _Float16* ceh = (_Float16*)ceH;
    for (int i = tid; i < NCONST * 32; i += 256) {
        int node = i >> 5, f = i & 31;
        float v = 0.f;
        if (f < 16)      v = (float)nh[node * 16 + f];
        else if (f < 22) v = ebar[node * 6 + (f - 16)];
        ceh[i] = (_Float16)v;
    }
    __syncthreads();

    // ======================= node stage: f_o (waves 0-3, 16 nodes each) =====
    if (w < 4) {
        ABReg B4, B5, B6, B7, B8, B9, B10, B11;
        #pragma unroll
        for (int v = 0; v < 8; ++v) {
            B4.u[v]  = Bimg[4  * 256 + v * 32 + lane];
            B5.u[v]  = Bimg[5  * 256 + v * 32 + lane];
            B6.u[v]  = Bimg[6  * 256 + v * 32 + lane];
            B7.u[v]  = Bimg[7  * 256 + v * 32 + lane];
            B8.u[v]  = Bimg[8  * 256 + v * 32 + lane];
            B9.u[v]  = Bimg[9  * 256 + v * 32 + lane];
            B10.u[v] = Bimg[10 * 256 + v * 32 + lane];
            B11.u[v] = Bimg[11 * 256 + v * 32 + lane];
        }
        const int node = w * 16 + m16;
        // ---- layer 1: 22 -> 45 (three N-tiles); scratch rows 64 halves ----
        ABReg a;
        #pragma unroll
        for (int v = 0; v < 4; ++v) {
            a.u[v]     = ceH[node * 16 + v + hi4];
            a.u[v + 4] = ceH[node * 16 + 8 + v + hi4];
        }
        v8f c0 = {}, c1 = {}, c2 = {};
        c0 = wmma16(a, B4, c0);
        c1 = wmma16(a, B5, c1);
        c2 = wmma16(a, B6, c2);
        const float bo10 = biasL[64 + m16], bo11 = biasL[64 + 16 + m16],
                    bo12 = biasL[64 + 32 + m16];
        #pragma unroll
        for (int v = 0; v < 8; ++v) {
            int M = v + hiM;
            sh[M * 64 + m16]      = (_Float16)qrelu8(c0[v] + bo10);
            sh[M * 64 + 16 + m16] = (_Float16)qrelu8(c1[v] + bo11);
            sh[M * 64 + 32 + m16] = (_Float16)qrelu8(c2[v] + bo12);
        }
        // ---- layer 2: 45 -> 22, chained K (two K-tiles x two N-tiles) ----
        ABReg a0, a1;
        #pragma unroll
        for (int v = 0; v < 4; ++v) {
            a0.u[v]     = shu[m16 * 32 + v + hi4];        // K 0..31
            a0.u[v + 4] = shu[m16 * 32 + 8 + v + hi4];
            a1.u[v]     = shu[m16 * 32 + 16 + v + hi4];   // K 32..63
            a1.u[v + 4] = shu[m16 * 32 + 24 + v + hi4];
        }
        v8f cn0 = {}, cn1 = {};
        cn0 = wmma16(a0, B7, cn0);
        cn0 = wmma16(a1, B9, cn0);
        cn1 = wmma16(a0, B8, cn1);
        cn1 = wmma16(a1, B10, cn1);
        const float bo20 = biasL[112 + m16], bo21 = biasL[112 + 16 + m16];
        #pragma unroll
        for (int v = 0; v < 8; ++v) {
            int M = v + hiM;
            sh[M * 64 + m16]      = (_Float16)qrelu8(cn0[v] + bo20);
            sh[M * 64 + 16 + m16] = (_Float16)qrelu8(cn1[v] + bo21);
        }
        // ---- layer 3: 22 -> 6, qrelu, reduce over nodes into dyn ----
        ABReg a3;
        #pragma unroll
        for (int v = 0; v < 4; ++v) {
            a3.u[v]     = shu[m16 * 32 + v + hi4];
            a3.u[v + 4] = shu[m16 * 32 + 8 + v + hi4];
        }
        v8f c3 = {};
        c3 = wmma16(a3, B11, c3);
        if (m16 < 6) {
            const float bo3 = biasL[144 + m16];
            float acc = 0.f;
            #pragma unroll
            for (int v = 0; v < 8; ++v) acc += qrelu8(c3[v] + bo3);
            atomicAdd(&dynL[m16], acc);
        }
    }
    __syncthreads();

    // ======================= f_c: 6 -> 48 -> 5, softmax ======================
    if (tid < 48) {
        float s = qb(fc_b1[tid]);
        #pragma unroll
        for (int i = 0; i < 6; ++i) s += dynL[i] * qb(fc_w1[i * 48 + tid]);
        afc[tid] = qrelu8(s);
    }
    __syncthreads();
    if (tid < 5) {
        float s = qb(fc_b2[tid]);
        for (int i = 0; i < 48; ++i) s += afc[i] * qb(fc_w2[i * 5 + tid]);
        logitsL[tid] = s;
    }
    __syncthreads();
    if (tid < 5) {
        float mx = logitsL[0];
        #pragma unroll
        for (int i = 1; i < 5; ++i) mx = fmaxf(mx, logitsL[i]);
        float den = 0.f;
        #pragma unroll
        for (int i = 0; i < 5; ++i) den += expf(logitsL[i] - mx);
        out[b * 5 + tid] = expf(logitsL[tid] - mx) / den;
    }
}

extern "C" void kernel_launch(void* const* d_in, const int* in_sizes, int n_in,
                              void* d_out, int out_size, void* d_ws, size_t ws_size,
                              hipStream_t stream) {
    (void)n_in; (void)out_size; (void)ws_size;
    const float* x        = (const float*)d_in[0];
    const float* bn_gamma = (const float*)d_in[1];
    const float* bn_beta  = (const float*)d_in[2];
    const float* bn_mean  = (const float*)d_in[3];
    const float* bn_var   = (const float*)d_in[4];
    const float* fr_w1    = (const float*)d_in[5];
    const float* fr_b1    = (const float*)d_in[6];
    const float* fr_w2    = (const float*)d_in[7];
    const float* fr_b2    = (const float*)d_in[8];
    const float* fr_w3    = (const float*)d_in[9];
    const float* fr_b3    = (const float*)d_in[10];
    const float* fo_w1    = (const float*)d_in[11];
    const float* fo_b1    = (const float*)d_in[12];
    const float* fo_w2    = (const float*)d_in[13];
    const float* fo_b2    = (const float*)d_in[14];
    const float* fo_w3    = (const float*)d_in[15];
    const float* fo_b3    = (const float*)d_in[16];
    const float* fc_w1    = (const float*)d_in[17];
    const float* fc_b1    = (const float*)d_in[18];
    const float* fc_w2    = (const float*)d_in[19];
    const float* fc_b2    = (const float*)d_in[20];

    unsigned* wsu = (unsigned*)d_ws;
    float*    wsf = (float*)d_ws;

    qin_prep<<<13, 256, 0, stream>>>(fr_w1, fr_w2, fr_w3, fo_w1, fo_w2, fo_w3,
                                     fr_b1, fr_b2, fr_b3, fo_b1, fo_b2, fo_b3,
                                     wsu, wsf);

    const int B = in_sizes[0] / (NCONST * NFEATS);   // 1024
    qin_main<<<B, 256, 0, stream>>>(x, bn_gamma, bn_beta, bn_mean, bn_var,
                                    fc_w1, fc_b1, fc_w2, fc_b2,
                                    wsu, wsf, (float*)d_out);
}
